// HeteroGAT_36747740184850
// MI455X (gfx1250) — matrix operand order repaired and checked by reading, not compiled, not run on hardware
//
#include <hip/hip_runtime.h>
#include <cstdint>

// ---------------------------------------------------------------------------
// HeteroGAT + HGT pipeline for MI455X (gfx1250, wave32, WMMA).
//
// Dense ops (all N x K @ K x 96 GEMMs, incl. block-diagonal per-head HGT
// transforms) run through one v_wmma_f32_16x16x32_f16 kernel (f16 in, f32
// accumulate), template-specialized on K (64/96) with M=96 compile-time so
// every address is base+immediate (independent load clauses, no pointer-chase
// chains), and 2 row-tiles register-blocked per wave (one B fragment feeds
// 2 WMMAs). Edge segment-softmax runs as flat atomic passes: ordered-uint
// atomicMax for segment_max, f32 atomicAdd for denominator / numerator,
// normalization deferred to a finalize pass (identical math to the reference
// since the denominator depends only on dst). All node tables (~170 MB) fit
// in the 192 MB global L2, so the scatter phase resolves at L2 bandwidth.
//
// Input layout assumption: harness flattens setup_inputs() dict in insertion
// order, recursively (lists in order, dict keys in insertion order):
//   0: x [N,64]
//   1..36:  gat[L][r]: W, att_src, att_dst, bias  (L=0..2, r=0..2)
//   37..42: bn[L]: gamma, beta (L=0..2)
//   43..46: proj[0].W, proj[0].b, proj[1].W, proj[1].b
//   47..58: hgt_mid:   Wk,bk,Wq,bq,Wv,bv,a_rel,m_rel,p_rel,Wa,ba,skip
//   59..70: hgt_final: same order
//   71..72: bn_mid gamma,beta   73..74: bn_final gamma,beta
//   75..76: lin.W, lin.b
//   77..79: edge_line, edge_region, edge_diag  (each int32 [2,E])
// Workspace use: ~168 MB (see offsets below).
// ---------------------------------------------------------------------------

typedef __attribute__((ext_vector_type(16))) _Float16 v16h;
typedef __attribute__((ext_vector_type(8)))  float    v8f;

#define HIDK 96
#define BN_EPS 1e-5f
#define NEG_INF_KEY 0x007FFFFFu   // f2ord(-inf)

__device__ __forceinline__ unsigned f2ord(float f) {
  unsigned b = __float_as_uint(f);
  return (b & 0x80000000u) ? ~b : (b | 0x80000000u);
}
__device__ __forceinline__ float ord2f(unsigned k) {
  unsigned b = (k & 0x80000000u) ? (k & 0x7FFFFFFFu) : ~k;
  return __uint_as_float(b);
}
__device__ __forceinline__ float lrelu(float x) { return x > 0.f ? x : 0.2f * x; }

// ---------------------------------------------------------------------------
// WMMA GEMM:  C[N x 96] = A[N x K] @ W[K x 96] (+ bias[96] if given)
// One wave per (RT x 16) x 16 output macro-tile; K and M compile-time so all
// strides are immediates. f32 -> f16 conversion on the fly; f32 accumulate.
// A fragment (16x32 f16): lanes 0-15 row M=l hold K {k0..k0+15},
// lanes 16-31 row M=l hold K {k0+16..k0+31} (matches B's k order).
// B fragment (32x16 f16): lanes 0-15 col N=l hold K {k0..k0+15}, lanes 16-31
// hold K {k0+16..k0+31}.
// ---------------------------------------------------------------------------
template <int K, int RT>
__global__ __launch_bounds__(128) void gemm_wmma_k(
    const float* __restrict__ A, const float* __restrict__ W,
    const float* __restrict__ bias, float* __restrict__ C, int N) {
  constexpr int M = HIDK;
  int wave   = blockIdx.x * blockDim.y + threadIdx.y;
  constexpr int tilesM = M / 16;
  int tilesN = (N + 16 * RT - 1) / (16 * RT);
  if (wave >= tilesM * tilesN) return;       // uniform per wave: EXEC stays full
  int tn = wave / tilesM, tm = wave - tn * tilesM;
  int row0 = tn * (16 * RT), col0 = tm << 4;

  int lane = threadIdx.x;
  int l    = lane & 15;
  bool hi  = lane > 15;
  int col  = col0 + l;

  const float* ap[RT];
#pragma unroll
  for (int rt = 0; rt < RT; ++rt) {
    int arow = row0 + rt * 16 + l;
    if (arow >= N) arow = N - 1;             // clamp (stores are guarded below)
    ap[rt] = A + (size_t)arow * K;
  }
  // B base for this lane: constexpr strides -> immediate offsets per element
  const float* wbase = W + (size_t)(hi ? 16 : 0) * M + col;

  v8f acc[RT];
#pragma unroll
  for (int rt = 0; rt < RT; ++rt) acc[rt] = (v8f){0.f,0.f,0.f,0.f,0.f,0.f,0.f,0.f};

#pragma unroll
  for (int k0 = 0; k0 < K; k0 += 32) {
    // B fragment: shared by all row tiles
    v16h bf;
#pragma unroll
    for (int j = 0; j < 16; ++j) bf[j] = (_Float16)wbase[(k0 + j) * M];

    int ka = k0 + (hi ? 16 : 0);
    int kb = ka + 8;
#pragma unroll
    for (int rt = 0; rt < RT; ++rt) {
      v16h af;
#pragma unroll
      for (int j = 0; j < 8; ++j) {
        af[j]     = (_Float16)ap[rt][ka + j];
        af[j + 8] = (_Float16)ap[rt][kb + j];
      }
      acc[rt] = __builtin_amdgcn_wmma_f32_16x16x32_f16(
          false, af, false, bf, (short)0, acc[rt], false, false);
    }
  }

  float bv = bias ? bias[col] : 0.f;
#pragma unroll
  for (int rt = 0; rt < RT; ++rt) {
    int rbase = row0 + rt * 16 + (hi ? 8 : 0);
    float* cp = C + (size_t)rbase * M + col;
#pragma unroll
    for (int r = 0; r < 8; ++r) {
      if (rbase + r < N) cp[r * M] = acc[rt][r] + bv;
    }
  }
}

// ------------------------------ utility fills ------------------------------
__global__ __launch_bounds__(256) void fill_f32_k(float* p, float v, int n) {
  int i = blockIdx.x * blockDim.x + threadIdx.x;
  if (i < n) p[i] = v;
}
__global__ __launch_bounds__(256) void fill_u32_k(unsigned* p, unsigned v, int n) {
  int i = blockIdx.x * blockDim.x + threadIdx.x;
  if (i < n) p[i] = v;
}
__global__ __launch_bounds__(256) void decode_m_k(const unsigned* __restrict__ mk,
                                                  float* __restrict__ m, int n) {
  int i = blockIdx.x * blockDim.x + threadIdx.x;
  if (i >= n) return;
  unsigned k = mk[i];
  m[i] = (k == NEG_INF_KEY) ? 0.f : ord2f(k);   // where(isfinite(m), m, 0)
}

// ------------------------------ GAT kernels --------------------------------
// a_s[n,h] = sum_d xp[n,h,d]*att_src[h,d]; same for a_d.
__global__ __launch_bounds__(256) void gat_coef_k(
    const float* __restrict__ xp, const float* __restrict__ asw,
    const float* __restrict__ adw, float* __restrict__ a_s,
    float* __restrict__ a_d, int N) {
  int t = blockIdx.x * blockDim.x + threadIdx.x;
  if (t >= N * 2) return;
  int n = t >> 1, h = t & 1;
  const float* row = xp + (size_t)n * HIDK + h * 48;
  float s = 0.f, d = 0.f;
#pragma unroll 8
  for (int j = 0; j < 48; ++j) {
    float v = row[j];
    s += v * asw[h * 48 + j];
    d += v * adw[h * 48 + j];
  }
  a_s[t] = s; a_d[t] = d;
}

__device__ __forceinline__ void edge_sd(const int* eg, int E, int e, int& s, int& d) {
  if (e < E) { s = eg[e]; d = eg[E + e]; }
  else       { s = d = e - E; }              // virtual self-loop
}

__global__ __launch_bounds__(256) void gat_max_k(
    const int* __restrict__ eg, int E, int nself,
    const float* __restrict__ a_s, const float* __restrict__ a_d,
    unsigned* __restrict__ mkey) {
  int t = blockIdx.x * blockDim.x + threadIdx.x;
  int tot = (E + nself) * 2;
  if (t >= tot) return;
  int e = t >> 1, h = t & 1, s, d;
  edge_sd(eg, E, e, s, d);
  float sc = lrelu(a_s[s * 2 + h] + a_d[d * 2 + h]);
  atomicMax(&mkey[d * 2 + h], f2ord(sc));
}

__global__ __launch_bounds__(256) void gat_den_k(
    const int* __restrict__ eg, int E, int nself,
    const float* __restrict__ a_s, const float* __restrict__ a_d,
    const float* __restrict__ m, float* __restrict__ den) {
  int t = blockIdx.x * blockDim.x + threadIdx.x;
  int tot = (E + nself) * 2;
  if (t >= tot) return;
  int e = t >> 1, h = t & 1, s, d;
  edge_sd(eg, E, e, s, d);
  float sc = lrelu(a_s[s * 2 + h] + a_d[d * 2 + h]);
  atomicAdd(&den[d * 2 + h], expf(sc - m[d * 2 + h]));
}

__global__ __launch_bounds__(256) void gat_num_k(
    const int* __restrict__ eg, int E, int nself,
    const float* __restrict__ a_s, const float* __restrict__ a_d,
    const float* __restrict__ m, const float* __restrict__ xp,
    float* __restrict__ num) {
  int t = blockIdx.x * blockDim.x + threadIdx.x;
  int tot = (E + nself) * HIDK;
  if (t >= tot) return;
  int e = t / HIDK, c = t - e * HIDK, h = c / 48, s, d;
  edge_sd(eg, E, e, s, d);
  float sc = lrelu(a_s[s * 2 + h] + a_d[d * 2 + h]);
  float ex = expf(sc - m[d * 2 + h]);
  atomicAdd(&num[(size_t)d * HIDK + c], ex * xp[(size_t)s * HIDK + c]);
}

// accum += num/(den+1e-16) + bias  (HeteroConv sum over relations)
__global__ __launch_bounds__(256) void gat_fin_k(
    const float* __restrict__ num, const float* __restrict__ den,
    const float* __restrict__ bias, float* __restrict__ accum, int N) {
  int t = blockIdx.x * blockDim.x + threadIdx.x;
  if (t >= N * HIDK) return;
  int n = t / HIDK, c = t - n * HIDK, h = c / 48;
  accum[t] += num[t] / (den[n * 2 + h] + 1e-16f) + bias[c];
}

// ------------------------------ BatchNorm ----------------------------------
__global__ __launch_bounds__(256) void bn_stats_k(const float* __restrict__ h,
                                                  float* __restrict__ stats, int N) {
  int c = blockIdx.x;
  float s = 0.f, s2 = 0.f;
  for (int n = threadIdx.x; n < N; n += blockDim.x) {
    float v = h[(size_t)n * HIDK + c];
    s += v; s2 += v * v;
  }
  __shared__ float sh[256], sh2[256];
  sh[threadIdx.x] = s; sh2[threadIdx.x] = s2;
  __syncthreads();
  for (int o = blockDim.x >> 1; o > 0; o >>= 1) {
    if ((int)threadIdx.x < o) {
      sh[threadIdx.x]  += sh[threadIdx.x + o];
      sh2[threadIdx.x] += sh2[threadIdx.x + o];
    }
    __syncthreads();
  }
  if (threadIdx.x == 0) {
    float mean = sh[0] / N;
    stats[c]        = mean;
    stats[HIDK + c] = sh2[0] / N - mean * mean;   // population var (ddof=0)
  }
}

// out = [lrelu]( gamma*(src-mean)*rsqrt(var+eps)+beta + add1 + add2 )
__global__ __launch_bounds__(256) void bn_apply_k(
    float* __restrict__ out, const float* __restrict__ src,
    const float* __restrict__ stats, const float* __restrict__ gamma,
    const float* __restrict__ beta, const float* __restrict__ add1,
    const float* __restrict__ add2, int doLrelu, int tot) {
  int t = blockIdx.x * blockDim.x + threadIdx.x;
  if (t >= tot) return;
  int c = t % HIDK;
  float y = gamma[c] * (src[t] - stats[c]) * rsqrtf(stats[HIDK + c] + BN_EPS) + beta[c];
  if (add1) y += add1[t];
  if (add2) y += add2[t];
  out[t] = doLrelu ? lrelu(y) : y;
}

// ------------------------------ HGT kernels --------------------------------
// Expand per-head 6x(16x16) matrices into a 96x96 block-diagonal weight.
__global__ __launch_bounds__(256) void blockdiag_k(const float* __restrict__ a,
                                                   float* __restrict__ w) {
  int t = blockIdx.x * blockDim.x + threadIdx.x;
  if (t >= 6 * 256) return;
  int h = t >> 8, idx = t & 255, d = idx >> 4, e = idx & 15;
  w[(h * 16 + d) * HIDK + h * 16 + e] = a[t];
}

// score[e,h] = 0.25 * p_rel[h] * dot(q[dst,h,:], kr[src,h,:]); atomicMax m.
__global__ __launch_bounds__(256) void hgt_score_k(
    const int* __restrict__ eg, int E, const float* __restrict__ q,
    const float* __restrict__ kr, const float* __restrict__ prel,
    float* __restrict__ score, unsigned* __restrict__ mkey) {
  int t = blockIdx.x * blockDim.x + threadIdx.x;
  if (t >= E * 6) return;
  int e = t / 6, h = t - e * 6;
  int s = eg[e], d = eg[E + e];
  const float* qp = q  + (size_t)d * HIDK + h * 16;
  const float* kp = kr + (size_t)s * HIDK + h * 16;
  float sc = 0.f;
#pragma unroll
  for (int j = 0; j < 16; ++j) sc += qp[j] * kp[j];
  sc *= prel[h] * 0.25f;                     // 1/sqrt(HD) = 0.25
  score[t] = sc;
  atomicMax(&mkey[d * 6 + h], f2ord(sc));
}

__global__ __launch_bounds__(256) void hgt_den_k(
    const int* __restrict__ eg, int E, const float* __restrict__ score,
    const float* __restrict__ m, float* __restrict__ den) {
  int t = blockIdx.x * blockDim.x + threadIdx.x;
  if (t >= E * 6) return;
  int e = t / 6, h = t - e * 6;
  int d = eg[E + e];
  atomicAdd(&den[d * 6 + h], expf(score[t] - m[d * 6 + h]));
}

__global__ __launch_bounds__(256) void hgt_agg_k(
    const int* __restrict__ eg, int E, const float* __restrict__ score,
    const float* __restrict__ m, const float* __restrict__ vr,
    float* __restrict__ agg) {
  int t = blockIdx.x * blockDim.x + threadIdx.x;
  if (t >= E * HIDK) return;
  int e = t / HIDK, c = t - e * HIDK, h = c >> 4;
  int s = eg[e], d = eg[E + e];
  float ex = expf(score[e * 6 + h] - m[d * 6 + h]);
  atomicAdd(&agg[(size_t)d * HIDK + c], ex * vr[(size_t)s * HIDK + c]);
}

// out = gelu(agg/(den+1e-16)), exact erf form
__global__ __launch_bounds__(256) void hgt_gelu_k(
    const float* __restrict__ agg, const float* __restrict__ den,
    float* __restrict__ out, int N) {
  int t = blockIdx.x * blockDim.x + threadIdx.x;
  if (t >= N * HIDK) return;
  int n = t / HIDK, c = t - n * HIDK, h = c >> 4;
  float x = agg[t] / (den[n * 6 + h] + 1e-16f);
  out[t] = 0.5f * x * (1.0f + erff(x * 0.70710678118654752f));
}

// g = s*g + (1-s)*h_in, s = sigmoid(skip)
__global__ __launch_bounds__(256) void hgt_skip_k(
    float* __restrict__ g, const float* __restrict__ hin,
    const float* __restrict__ skip, int tot) {
  int t = blockIdx.x * blockDim.x + threadIdx.x;
  if (t >= tot) return;
  float s = 1.0f / (1.0f + expf(-skip[0]));
  g[t] = s * g[t] + (1.0f - s) * hin[t];
}

// logits[n] = dot(h[n,:], W) + b
__global__ __launch_bounds__(256) void lin_k(
    const float* __restrict__ h, const float* __restrict__ W,
    const float* __restrict__ b, float* __restrict__ out, int N) {
  int n = blockIdx.x * blockDim.x + threadIdx.x;
  if (n >= N) return;
  float s = b[0];
  const float* row = h + (size_t)n * HIDK;
#pragma unroll 8
  for (int c = 0; c < HIDK; ++c) s += row[c] * W[c];
  out[n] = s;
}

// ---------------------------------------------------------------------------
extern "C" void kernel_launch(void* const* d_in, const int* in_sizes, int n_in,
                              void* d_out, int out_size, void* d_ws, size_t ws_size,
                              hipStream_t stream) {
  (void)ws_size; (void)out_size;
  const int IN = 64;
  const int N  = in_sizes[0] / IN;
  const int E  = in_sizes[n_in - 3] / 2;

  auto PF = [&](int i) { return (const float*)d_in[i]; };
  const float* x = PF(0);
  const int* edges[3] = { (const int*)d_in[n_in - 3],
                          (const int*)d_in[n_in - 2],
                          (const int*)d_in[n_in - 1] };

  // ---- workspace layout (floats) ----
  float* wsf = (float*)d_ws;
  size_t off = 0;
  auto alloc = [&](size_t n) { float* p = wsf + off; off += n; return p; };
  const size_t NS = (size_t)N * HIDK;
  float* h    = alloc(NS);        // node state
  float* t0   = alloc(NS);        // xp / kr / vr / gelu buffer
  float* t1   = alloc(NS);        // num / agg / proj buffer
  float* t2   = alloc(NS);        // relation accumulator / hgt output
  float* Kb   = alloc(NS);
  float* Qb   = alloc(NS);
  float* Vb   = alloc(NS);
  float* sc[3]; for (int r = 0; r < 3; ++r) sc[r] = alloc((size_t)E * 6);
  float* a_s  = alloc((size_t)N * 2);
  float* a_d  = alloc((size_t)N * 2);
  float* den  = alloc((size_t)N * 6);
  unsigned* mkey = (unsigned*)alloc((size_t)N * 6);
  float* mval = alloc((size_t)N * 6);
  float* stats = alloc(2 * HIDK);
  float* wblk  = alloc(HIDK * HIDK);

  auto nb = [](long long n, int b) { return (unsigned)((n + b - 1) / b); };
  auto fillf = [&](float* p, float v, long long n) {
    fill_f32_k<<<nb(n, 256), 256, 0, stream>>>(p, v, (int)n);
  };
  auto fillu = [&](unsigned* p, unsigned v, long long n) {
    fill_u32_k<<<nb(n, 256), 256, 0, stream>>>(p, v, (int)n);
  };
  // K is always 64 or 96 in this workload; M fixed at 96; RT=2 row-tiles/wave.
  auto gemm = [&](const float* A, const float* W, const float* bias, float* C,
                  int n, int k) {
    int tiles = ((n + 31) / 32) * (HIDK / 16);
    dim3 blk(32, 4);
    if (k == 64) {
      gemm_wmma_k<64, 2><<<nb(tiles, 4), blk, 0, stream>>>(A, W, bias, C, n);
    } else {
      gemm_wmma_k<96, 2><<<nb(tiles, 4), blk, 0, stream>>>(A, W, bias, C, n);
    }
  };

  // ---- one GAT layer (HeteroConv sum over 3 relations), accum -> t2 ----
  auto gat_layer = [&](const float* inA, int K, int L) {
    fillf(t2, 0.f, NS);
    for (int r = 0; r < 3; ++r) {
      int gb = 1 + (L * 3 + r) * 4;   // W, att_src, att_dst, bias
      gemm(inA, PF(gb + 0), nullptr, t0, N, K);                // xp = in @ W
      gat_coef_k<<<nb((long long)N * 2, 256), 256, 0, stream>>>(
          t0, PF(gb + 1), PF(gb + 2), a_s, a_d, N);
      fillu(mkey, NEG_INF_KEY, (long long)N * 2);
      fillf(den, 0.f, (long long)N * 2);
      fillf(t1, 0.f, NS);
      long long tot2 = (long long)(E + N) * 2;
      long long totc = (long long)(E + N) * HIDK;
      gat_max_k<<<nb(tot2, 256), 256, 0, stream>>>(edges[r], E, N, a_s, a_d, mkey);
      decode_m_k<<<nb((long long)N * 2, 256), 256, 0, stream>>>(mkey, mval, N * 2);
      gat_den_k<<<nb(tot2, 256), 256, 0, stream>>>(edges[r], E, N, a_s, a_d, mval, den);
      gat_num_k<<<nb(totc, 256), 256, 0, stream>>>(edges[r], E, N, a_s, a_d, mval, t0, t1);
      gat_fin_k<<<nb(NS, 256), 256, 0, stream>>>(t1, den, PF(gb + 3), t2, N);
    }
  };

  // ---- BatchNorm (+ optional residual adds, + optional lrelu) ----
  auto bn_do = [&](float* out, const float* src, const float* gamma,
                   const float* beta, const float* add1, const float* add2,
                   int doLrelu) {
    bn_stats_k<<<HIDK, 256, 0, stream>>>(src, stats, N);
    bn_apply_k<<<nb(NS, 256), 256, 0, stream>>>(out, src, stats, gamma, beta,
                                                add1, add2, doLrelu, (int)NS);
  };

  // ---- one HGT block, reads h, writes t2 (pre-BN output g-candidate) ----
  auto hgt_block = [&](int base) {
    const float *Wk = PF(base + 0), *bk = PF(base + 1), *Wq = PF(base + 2),
                *bq = PF(base + 3), *Wv = PF(base + 4), *bv = PF(base + 5),
                *a_rel = PF(base + 6), *m_rel = PF(base + 7),
                *p_rel = PF(base + 8), *Wa = PF(base + 9), *ba = PF(base + 10),
                *skip = PF(base + 11);
    gemm(h, Wk, bk, Kb, N, HIDK);
    gemm(h, Wq, bq, Qb, N, HIDK);
    gemm(h, Wv, bv, Vb, N, HIDK);
    fillu(mkey, NEG_INF_KEY, (long long)N * 6);
    fillf(den, 0.f, (long long)N * 6);
    fillf(t1, 0.f, NS);
    long long tot6 = (long long)E * 6;
    long long totc = (long long)E * HIDK;
    for (int r = 0; r < 3; ++r) {            // scores + global max
      fillf(wblk, 0.f, HIDK * HIDK);
      blockdiag_k<<<nb(6 * 256, 256), 256, 0, stream>>>(a_rel + r * 1536, wblk);
      gemm(Kb, wblk, nullptr, t0, N, HIDK);          // kr
      hgt_score_k<<<nb(tot6, 256), 256, 0, stream>>>(edges[r], E, Qb, t0,
                                                     p_rel + r * 6, sc[r], mkey);
    }
    decode_m_k<<<nb((long long)N * 6, 256), 256, 0, stream>>>(mkey, mval, N * 6);
    for (int r = 0; r < 3; ++r) {            // denominator + aggregation
      fillf(wblk, 0.f, HIDK * HIDK);
      blockdiag_k<<<nb(6 * 256, 256), 256, 0, stream>>>(m_rel + r * 1536, wblk);
      gemm(Vb, wblk, nullptr, t0, N, HIDK);          // vr
      hgt_den_k<<<nb(tot6, 256), 256, 0, stream>>>(edges[r], E, sc[r], mval, den);
      hgt_agg_k<<<nb(totc, 256), 256, 0, stream>>>(edges[r], E, sc[r], mval, t0, t1);
    }
    hgt_gelu_k<<<nb(NS, 256), 256, 0, stream>>>(t1, den, t0, N);
    gemm(t0, Wa, ba, t2, N, HIDK);
    hgt_skip_k<<<nb(NS, 256), 256, 0, stream>>>(t2, h, skip, (int)NS);
  };

  // ============================ pipeline ============================
  // Layer 0: h = lrelu(bn0(sum_r gat(x)))
  gat_layer(x, IN, 0);
  bn_do(h, t2, PF(37), PF(38), nullptr, nullptr, 1);

  // Layer 1: h = lrelu(h + bn1(gat(h)) + x@proj0)
  gat_layer(h, HIDK, 1);
  gemm(x, PF(43), PF(44), t1, N, IN);
  bn_do(h, t2, PF(39), PF(40), h, t1, 1);

  // mid HGT: h = lrelu(h + bn_mid(hgt_mid(h)))
  hgt_block(47);
  bn_do(h, t2, PF(71), PF(72), h, nullptr, 1);

  // Layer 2: h = lrelu(h + bn2(gat(h)) + x@proj1)
  gat_layer(h, HIDK, 2);
  gemm(x, PF(45), PF(46), t1, N, IN);
  bn_do(h, t2, PF(41), PF(42), h, t1, 1);

  // final HGT: h = lrelu(h + bn_final(hgt_final(h)))
  hgt_block(59);
  bn_do(h, t2, PF(73), PF(74), h, nullptr, 1);

  // logits
  lin_k<<<nb(N, 256), 256, 0, stream>>>(h, PF(75), PF(76), (float*)d_out, N);
}
